// RandLANet_23484881174680
// MI455X (gfx1250) — compile-verified, matrix operand order checked by reading, not run on hardware
//
#include <hip/hip_runtime.h>
#include <math.h>

typedef __attribute__((ext_vector_type(16))) _Float16 v16h;
typedef __attribute__((ext_vector_type(8)))  _Float16 v8h;
typedef __attribute__((ext_vector_type(8)))  float    v8f;

#define KSTEP 32
#define LDSTRIDE 40   // halves; 80 bytes, 16B aligned, conflict-free b128 reads

// ---------------------------------------------------------------------------
// Pack f32 weights (O x C) into zero-padded f16 (O x Cpad), Cpad % 32 == 0.
// ---------------------------------------------------------------------------
__global__ void pack_weights(const float* __restrict__ W, _Float16* __restrict__ Wpk,
                             int O, int C, int Cpad)
{
    int t = blockIdx.x * blockDim.x + threadIdx.x;
    if (t >= O * Cpad) return;
    int r = t / Cpad, c = t - r * Cpad;
    Wpk[t] = (c < C) ? (_Float16)W[(size_t)r * C + c] : (_Float16)0.0f;
}

// ---------------------------------------------------------------------------
// Generic WMMA GEMM:  Y = act( Wpk * X )
//   Wpk : O x Cpad f16 (zero-padded), O % 64 == 0, Cpad % 32 == 0
//   X   : Cpad x M f16 row-major (pad rows zeroed by producer), M % 32 == 0
//   epilogue: v = v*gamma[row] + beta[row] (each optional); leaky 0.1 optional
//   outputs: Y32 (f32 OxM) and/or Y16 (f16 OxM)
// Block: 128 threads = 4 waves; tile 64 rows x 32 cols; k-step 32.
// Unconditional vectorized staging; fragments via 2x ds_load_b128 each
// (ISA 16-bit layout: lane m=lane&15, kbase=(lane<16)?0:8, contiguous 16B
//  runs at half-offsets kbase and kbase+16).
// ---------------------------------------------------------------------------
__global__ __launch_bounds__(128)
void wmma_gemm_bnact(const _Float16* __restrict__ Wpk, const _Float16* __restrict__ X,
                     const float* __restrict__ gamma, const float* __restrict__ beta,
                     float* __restrict__ Y32, _Float16* __restrict__ Y16,
                     int Cpad, int M, int leaky)
{
    __shared__ __align__(16) _Float16 Wt[64][LDSTRIDE];    // [row][k]
    __shared__ __align__(16) _Float16 XtT[32][LDSTRIDE];   // [col][k]  (k-contiguous)

    const int tid  = threadIdx.x;
    const int wave = tid >> 5;
    const int lane = tid & 31;
    const int rowBase = blockIdx.y * 64;
    const int colBase = blockIdx.x * 32;

    v8f acc0 = {0.f, 0.f, 0.f, 0.f, 0.f, 0.f, 0.f, 0.f};
    v8f acc1 = {0.f, 0.f, 0.f, 0.f, 0.f, 0.f, 0.f, 0.f};

    const int mrow  = lane & 15;
    const int kbase = (lane < 16) ? 0 : 8;

    // staging coordinates (8-half chunks)
    const int xk = tid >> 2;              // 0..31  (k row of X tile)
    const int xc = (tid & 3) * 8;         // 0,8,16,24 (col chunk)

    for (int k0 = 0; k0 < Cpad; k0 += KSTEP) {
        // stage W tile (64 x 32): 2 x v8h per thread, no guards
        #pragma unroll
        for (int h = 0; h < 2; ++h) {
            int ii = tid + h * 128;       // 0..255 chunks
            int r = ii >> 2, cc = (ii & 3) * 8;
            *(v8h*)&Wt[r][cc] =
                *(const v8h*)&Wpk[(size_t)(rowBase + r) * Cpad + (k0 + cc)];
        }
        // stage X tile transposed: coalesced v8h load + 8 b16 LDS scatter
        {
            v8h xv = *(const v8h*)&X[(size_t)(k0 + xk) * M + (colBase + xc)];
            #pragma unroll
            for (int j = 0; j < 8; ++j) XtT[xc + j][xk] = xv[j];
        }
        __syncthreads();

        const v8h* pa  = (const v8h*)&Wt[wave * 16 + mrow][kbase];
        const v8h* pb0 = (const v8h*)&XtT[mrow][kbase];
        const v8h* pb1 = (const v8h*)&XtT[16 + mrow][kbase];
        v16h a  = __builtin_shufflevector(pa[0],  pa[2],
                    0,1,2,3,4,5,6,7,8,9,10,11,12,13,14,15);
        v16h b0 = __builtin_shufflevector(pb0[0], pb0[2],
                    0,1,2,3,4,5,6,7,8,9,10,11,12,13,14,15);
        v16h b1 = __builtin_shufflevector(pb1[0], pb1[2],
                    0,1,2,3,4,5,6,7,8,9,10,11,12,13,14,15);

        acc0 = __builtin_amdgcn_wmma_f32_16x16x32_f16(false, a, false, b0,
                                                      (short)0, acc0, false, false);
        acc1 = __builtin_amdgcn_wmma_f32_16x16x32_f16(false, a, false, b1,
                                                      (short)0, acc1, false, false);
        __syncthreads();
    }

    // D layout: lane 0-15 -> N=lane, M=vgpr; lane 16-31 -> N=lane-16, M=vgpr+8
    const int c0   = colBase + (lane & 15);
    const int rsub = (lane < 16) ? 0 : 8;
    #pragma unroll
    for (int r = 0; r < 8; ++r) {
        int row = rowBase + wave * 16 + rsub + r;
        float v0 = acc0[r], v1 = acc1[r];
        if (gamma) { float g = gamma[row]; v0 *= g; v1 *= g; }
        if (beta)  { float bb = beta[row]; v0 += bb; v1 += bb; }
        if (leaky) { v0 = (v0 > 0.f) ? v0 : 0.1f * v0;
                     v1 = (v1 > 0.f) ? v1 : 0.1f * v1; }
        size_t o0 = (size_t)row * M + c0;
        if (Y32) { Y32[o0] = v0; Y32[o0 + 16] = v1; }
        if (Y16) { Y16[o0] = (_Float16)v0; Y16[o0 + 16] = (_Float16)v1; }
    }
}

// ---------------------------------------------------------------------------
// Brute-force KNN (k=16): one query per thread, sources tiled through LDS,
// fully unrolled predicated bubble-insert. Ascending distance, ties keep
// lower index (matches top_k(-d2)).
// ---------------------------------------------------------------------------
__global__ __launch_bounds__(128)
void knn16_kernel(const float* __restrict__ xyz, int N, int* __restrict__ out)
{
    __shared__ float sx[128], sy[128], sz[128];
    const int q = blockIdx.x * 128 + threadIdx.x;
    float qx = 0.f, qy = 0.f, qz = 0.f;
    if (q < N) { qx = xyz[3*q]; qy = xyz[3*q+1]; qz = xyz[3*q+2]; }

    float bd[16]; int bi[16];
    #pragma unroll
    for (int t = 0; t < 16; ++t) { bd[t] = 3.4e38f; bi[t] = 0; }

    for (int s0 = 0; s0 < N; s0 += 128) {
        __syncthreads();
        int s = s0 + threadIdx.x;
        if (s < N) { sx[threadIdx.x] = xyz[3*s]; sy[threadIdx.x] = xyz[3*s+1]; sz[threadIdx.x] = xyz[3*s+2]; }
        __syncthreads();
        int lim = N - s0; if (lim > 128) lim = 128;
        for (int j = 0; j < lim; ++j) {
            float dx = qx - sx[j], dy = qy - sy[j], dz = qz - sz[j];
            float d  = dx*dx + dy*dy + dz*dz;
            int   id = s0 + j;
            #pragma unroll
            for (int t = 0; t < 16; ++t) {
                if (d < bd[t]) {
                    float td = bd[t]; int ti = bi[t];
                    bd[t] = d; bi[t] = id; d = td; id = ti;
                }
            }
        }
    }
    if (q < N) {
        #pragma unroll
        for (int t = 0; t < 16; ++t) out[q*16 + t] = bi[t];
    }
}

// Nearest source index per target point (argmin of pdist2; first-min tie).
__global__ __launch_bounds__(128)
void nn1_kernel(const float* __restrict__ tgt, int Nt,
                const float* __restrict__ src, int Ns, int* __restrict__ out)
{
    __shared__ float sx[128], sy[128], sz[128];
    const int q = blockIdx.x * 128 + threadIdx.x;
    float qx = 0.f, qy = 0.f, qz = 0.f;
    if (q < Nt) { qx = tgt[3*q]; qy = tgt[3*q+1]; qz = tgt[3*q+2]; }
    float best = 3.4e38f; int bi = 0;
    for (int s0 = 0; s0 < Ns; s0 += 128) {
        __syncthreads();
        int s = s0 + threadIdx.x;
        if (s < Ns) { sx[threadIdx.x] = src[3*s]; sy[threadIdx.x] = src[3*s+1]; sz[threadIdx.x] = src[3*s+2]; }
        __syncthreads();
        int lim = Ns - s0; if (lim > 128) lim = 128;
        for (int j = 0; j < lim; ++j) {
            float dx = qx - sx[j], dy = qy - sy[j], dz = qz - sz[j];
            float d  = dx*dx + dy*dy + dz*dz;
            if (d < best) { best = d; bi = s0 + j; }
        }
    }
    if (q < Nt) out[q] = bi;
}

// ---------------------------------------------------------------------------
// LFA input assembly: Xo[Cpad x (Np*16)] f16, column = n*16+k
// rows [0,C): gathered neighbor features; rows C..C+9: diff(3),dist,center(3),
// neigh(3); rows C+10..Cpad-1 zeroed (GEMM pad).
// ---------------------------------------------------------------------------
__global__ void lfa_assemble(const float* __restrict__ xyz, const float* __restrict__ feat,
                             const int* __restrict__ idx, _Float16* __restrict__ Xo,
                             int Np, int C, int Cpad)
{
    const int M = Np * 16;
    int t = blockIdx.x * blockDim.x + threadIdx.x;
    if (t >= M) return;
    int n  = t >> 4;
    int nb = idx[t];
    float cx = xyz[3*n],  cy = xyz[3*n+1],  cz = xyz[3*n+2];
    float nx = xyz[3*nb], ny = xyz[3*nb+1], nz = xyz[3*nb+2];
    float dx = cx - nx, dy = cy - ny, dz = cz - nz;
    float dist = sqrtf(dx*dx + dy*dy + dz*dz);
    for (int c = 0; c < C; ++c)
        Xo[(size_t)c * M + t] = (_Float16)feat[(size_t)c * Np + nb];
    _Float16* p = Xo + (size_t)C * M + t;
    size_t m = (size_t)M;
    p[0*m] = (_Float16)dx;   p[1*m] = (_Float16)dy;   p[2*m] = (_Float16)dz;
    p[3*m] = (_Float16)dist; p[4*m] = (_Float16)cx;   p[5*m] = (_Float16)cy;
    p[6*m] = (_Float16)cz;   p[7*m] = (_Float16)nx;   p[8*m] = (_Float16)ny;
    p[9*m] = (_Float16)nz;
    for (int c = C + 10; c < Cpad; ++c)
        Xo[(size_t)c * M + t] = (_Float16)0.0f;
}

// softmax over K=16 neighbors (per output channel) + weighted sum of values
__global__ void attn_pool(const float* __restrict__ L, const _Float16* __restrict__ V,
                          _Float16* __restrict__ P, int O, int Np)
{
    int t = blockIdx.x * blockDim.x + threadIdx.x;
    if (t >= O * Np) return;
    int o = t / Np, n = t - o * Np;
    size_t base = (size_t)o * Np * 16 + (size_t)n * 16;
    float mx = -3.4e38f;
    #pragma unroll
    for (int k = 0; k < 16; ++k) mx = fmaxf(mx, L[base + k]);
    float w[16]; float s = 0.f;
    #pragma unroll
    for (int k = 0; k < 16; ++k) { w[k] = __expf(L[base + k] - mx); s += w[k]; }
    float inv = 1.0f / s;
    float accv = 0.f;
    #pragma unroll
    for (int k = 0; k < 16; ++k) accv += w[k] * inv * (float)V[base + k];
    P[(size_t)o * Np + n] = (_Float16)accv;
}

__global__ void gather_feat(const float* __restrict__ F, const int* __restrict__ sel,
                            float* __restrict__ out, int C, int Nsrc, int Mdst)
{
    int t = blockIdx.x * blockDim.x + threadIdx.x;
    if (t >= C * Mdst) return;
    int c = t / Mdst, m = t - c * Mdst;
    out[(size_t)c * Mdst + m] = F[(size_t)c * Nsrc + sel[m]];
}

__global__ void gather_xyz(const float* __restrict__ xyz, const int* __restrict__ sel,
                           float* __restrict__ out, int M)
{
    int m = blockIdx.x * blockDim.x + threadIdx.x;
    if (m >= M) return;
    int s = sel[m];
    out[3*m] = xyz[3*s]; out[3*m+1] = xyz[3*s+1]; out[3*m+2] = xyz[3*s+2];
}

// rows [0,Ca): A[:, nn[n]] (nearest-neighbor interp); rows [Ca,Ca+Cb): B[:, n]
__global__ void concat_interp(const float* __restrict__ A, const int* __restrict__ nn,
                              const float* __restrict__ Bf, _Float16* __restrict__ Xo,
                              int Ca, int Cb, int Nt, int Ns)
{
    int t = blockIdx.x * blockDim.x + threadIdx.x;
    if (t >= (Ca + Cb) * Nt) return;
    int r = t / Nt, n = t - r * Nt;
    float v = (r < Ca) ? A[(size_t)r * Ns + nn[n]] : Bf[(size_t)(r - Ca) * Nt + n];
    Xo[(size_t)t] = (_Float16)v;
}

// (N,6) f32 -> (32,N) f16 with rows 6..31 zeroed (GEMM pad)
__global__ void transpose_feat(const float* __restrict__ feat, _Float16* __restrict__ Xo, int N)
{
    int n = blockIdx.x * blockDim.x + threadIdx.x;
    if (n >= N) return;
    for (int c = 0; c < 6; ++c)  Xo[(size_t)c * N + n] = (_Float16)feat[(size_t)n * 6 + c];
    for (int c = 6; c < 32; ++c) Xo[(size_t)c * N + n] = (_Float16)0.0f;
}

// final head: logits[n][o] = Wh2(13x64) * h(64xN) + bh2, written transposed (N,13)
__global__ void head2_kernel(const _Float16* __restrict__ h, const float* __restrict__ Wh2,
                             const float* __restrict__ bh2, float* __restrict__ out, int N)
{
    int n = blockIdx.x * blockDim.x + threadIdx.x;
    if (n >= N) return;
    for (int o = 0; o < 13; ++o) {
        float s = bh2[o];
        for (int c = 0; c < 64; ++c)
            s += Wh2[o * 64 + c] * (float)h[(size_t)c * N + n];
        out[(size_t)n * 13 + o] = s;
    }
}

// ---------------------------------------------------------------------------
extern "C" void kernel_launch(void* const* d_in, const int* in_sizes, int n_in,
                              void* d_out, int out_size, void* d_ws, size_t ws_size,
                              hipStream_t stream) {
    (void)in_sizes; (void)n_in; (void)out_size; (void)ws_size;

    const int B = 2, N = 8192, M1 = 4096, M2 = 2048;
    const int NK0 = N * 16, NK1 = M1 * 16, NK2 = M2 * 16;

    const float* xyz  = (const float*)d_in[0];
    const float* feat = (const float*)d_in[1];
    const int*   sel1 = (const int*)d_in[2];
    const int*   sel2 = (const int*)d_in[3];
    const float* Wpre = (const float*)d_in[4];
    const float* gpre = (const float*)d_in[5];
    const float* bpre = (const float*)d_in[6];
    const float* W1m = (const float*)d_in[7],  *g1m = (const float*)d_in[8],  *b1m = (const float*)d_in[9];
    const float* W1s = (const float*)d_in[10], *W1p = (const float*)d_in[11];
    const float* g1p = (const float*)d_in[12], *b1p = (const float*)d_in[13];
    const float* W2m = (const float*)d_in[14], *g2m = (const float*)d_in[15], *b2m = (const float*)d_in[16];
    const float* W2s = (const float*)d_in[17], *W2p = (const float*)d_in[18];
    const float* g2p = (const float*)d_in[19], *b2p = (const float*)d_in[20];
    const float* W3m = (const float*)d_in[21], *g3m = (const float*)d_in[22], *b3m = (const float*)d_in[23];
    const float* W3s = (const float*)d_in[24], *W3p = (const float*)d_in[25];
    const float* g3p = (const float*)d_in[26], *b3p = (const float*)d_in[27];
    const float* Wu2 = (const float*)d_in[28], *gu2 = (const float*)d_in[29], *bu2 = (const float*)d_in[30];
    const float* Wu1 = (const float*)d_in[31], *gu1 = (const float*)d_in[32], *bu1 = (const float*)d_in[33];
    const float* Wh1 = (const float*)d_in[34], *bh1 = (const float*)d_in[35];
    const float* Wh2 = (const float*)d_in[36], *bh2 = (const float*)d_in[37];
    float* out = (float*)d_out;

    char* ws = (char*)d_ws;
    size_t off = 0;
    auto alloc = [&](size_t bytes) -> char* {
        char* p = ws + off;
        off = (off + bytes + 255) & ~(size_t)255;
        return p;
    };
    int*      idx0  = (int*)alloc((size_t)N  * 16 * 4);
    int*      idx1  = (int*)alloc((size_t)M1 * 16 * 4);
    int*      idx2  = (int*)alloc((size_t)M2 * 16 * 4);
    int*      nn21  = (int*)alloc((size_t)M1 * 4);
    int*      nn10  = (int*)alloc((size_t)N * 4);
    float*    xyz1  = (float*)alloc((size_t)M1 * 3 * 4);
    float*    xyz2  = (float*)alloc((size_t)M2 * 3 * 4);
    float*    x0    = (float*)alloc((size_t)64  * N  * 4);
    float*    x1    = (float*)alloc((size_t)64  * N  * 4);
    float*    f1    = (float*)alloc((size_t)64  * M1 * 4);
    float*    x2    = (float*)alloc((size_t)128 * M1 * 4);
    float*    f2    = (float*)alloc((size_t)128 * M2 * 4);
    float*    x3    = (float*)alloc((size_t)256 * M2 * 4);
    float*    d2    = (float*)alloc((size_t)128 * M1 * 4);
    _Float16* d1h   = (_Float16*)alloc((size_t)64 * N * 2);
    _Float16* hh    = (_Float16*)alloc((size_t)64 * N * 2);
    _Float16* pooled= (_Float16*)alloc((size_t)64 * N * 2);       // max O*Np = 524288
    _Float16* wpk   = (_Float16*)alloc((size_t)256 * 384 * 2);    // packed weights
    _Float16* bigX  = (_Float16*)alloc((size_t)96 * NK0 * 2);     // assemble (padded rows)
    _Float16* bigV  = (_Float16*)alloc((size_t)64 * NK0 * 2);     // bn_act values f16
    float*    bigL  = (float*)alloc((size_t)64 * NK0 * 4);        // attention logits f32

    auto cdiv = [](int a, int b) { return (a + b - 1) / b; };
    auto pack = [&](const float* W, int O, int C, int Cpad) {
        pack_weights<<<cdiv(O * Cpad, 256), 256, 0, stream>>>(W, wpk, O, C, Cpad);
    };

    for (int b = 0; b < B; ++b) {
        const float* xyzb  = xyz  + (size_t)b * N * 3;
        const float* featb = feat + (size_t)b * N * 6;
        const int*   sel1b = sel1 + (size_t)b * M1;
        const int*   sel2b = sel2 + (size_t)b * M2;
        float*       outb  = out  + (size_t)b * N * 13;

        // ---- pre: x0 = bn_act(Wpre . feat^T) ----
        transpose_feat<<<cdiv(N, 256), 256, 0, stream>>>(featb, bigX, N);
        pack(Wpre, 64, 6, 32);
        wmma_gemm_bnact<<<dim3(N / 32, 1), 128, 0, stream>>>(
            wpk, bigX, gpre, bpre, x0, nullptr, 32, N, 1);

        // ---- level 1 (N=8192, C=64 -> O=64) ----
        knn16_kernel<<<N / 128, 128, 0, stream>>>(xyzb, N, idx0);
        lfa_assemble<<<cdiv(NK0, 256), 256, 0, stream>>>(xyzb, x0, idx0, bigX, N, 64, 96);
        pack(W1m, 64, 74, 96);
        wmma_gemm_bnact<<<dim3(NK0 / 32, 1), 128, 0, stream>>>(
            wpk, bigX, g1m, b1m, nullptr, bigV, 96, NK0, 1);
        pack(W1s, 64, 64, 64);
        wmma_gemm_bnact<<<dim3(NK0 / 32, 1), 128, 0, stream>>>(
            wpk, bigV, nullptr, nullptr, bigL, nullptr, 64, NK0, 0);
        attn_pool<<<cdiv(64 * N, 256), 256, 0, stream>>>(bigL, bigV, pooled, 64, N);
        pack(W1p, 64, 64, 64);
        wmma_gemm_bnact<<<dim3(N / 32, 1), 128, 0, stream>>>(
            wpk, pooled, g1p, b1p, x1, nullptr, 64, N, 1);

        // ---- level 2 (M1=4096, C=64 -> O=128) ----
        gather_xyz<<<cdiv(M1, 256), 256, 0, stream>>>(xyzb, sel1b, xyz1, M1);
        gather_feat<<<cdiv(64 * M1, 256), 256, 0, stream>>>(x1, sel1b, f1, 64, N, M1);
        knn16_kernel<<<M1 / 128, 128, 0, stream>>>(xyz1, M1, idx1);
        lfa_assemble<<<cdiv(NK1, 256), 256, 0, stream>>>(xyz1, f1, idx1, bigX, M1, 64, 96);
        pack(W2m, 128, 74, 96);
        wmma_gemm_bnact<<<dim3(NK1 / 32, 2), 128, 0, stream>>>(
            wpk, bigX, g2m, b2m, nullptr, bigV, 96, NK1, 1);
        pack(W2s, 128, 128, 128);
        wmma_gemm_bnact<<<dim3(NK1 / 32, 2), 128, 0, stream>>>(
            wpk, bigV, nullptr, nullptr, bigL, nullptr, 128, NK1, 0);
        attn_pool<<<cdiv(128 * M1, 256), 256, 0, stream>>>(bigL, bigV, pooled, 128, M1);
        pack(W2p, 128, 128, 128);
        wmma_gemm_bnact<<<dim3(M1 / 32, 2), 128, 0, stream>>>(
            wpk, pooled, g2p, b2p, x2, nullptr, 128, M1, 1);

        // ---- level 3 (M2=2048, C=128 -> O=256) ----
        gather_xyz<<<cdiv(M2, 256), 256, 0, stream>>>(xyz1, sel2b, xyz2, M2);
        gather_feat<<<cdiv(128 * M2, 256), 256, 0, stream>>>(x2, sel2b, f2, 128, M1, M2);
        knn16_kernel<<<M2 / 128, 128, 0, stream>>>(xyz2, M2, idx2);
        lfa_assemble<<<cdiv(NK2, 256), 256, 0, stream>>>(xyz2, f2, idx2, bigX, M2, 128, 160);
        pack(W3m, 256, 138, 160);
        wmma_gemm_bnact<<<dim3(NK2 / 32, 4), 128, 0, stream>>>(
            wpk, bigX, g3m, b3m, nullptr, bigV, 160, NK2, 1);
        pack(W3s, 256, 256, 256);
        wmma_gemm_bnact<<<dim3(NK2 / 32, 4), 128, 0, stream>>>(
            wpk, bigV, nullptr, nullptr, bigL, nullptr, 256, NK2, 0);
        attn_pool<<<cdiv(256 * M2, 256), 256, 0, stream>>>(bigL, bigV, pooled, 256, M2);
        pack(W3p, 256, 256, 256);
        wmma_gemm_bnact<<<dim3(M2 / 32, 4), 128, 0, stream>>>(
            wpk, pooled, g3p, b3p, x3, nullptr, 256, M2, 1);

        // ---- decoder ----
        nn1_kernel<<<cdiv(M1, 128), 128, 0, stream>>>(xyz1, M1, xyz2, M2, nn21);
        concat_interp<<<cdiv(384 * M1, 256), 256, 0, stream>>>(x3, nn21, x2, bigX, 256, 128, M1, M2);
        pack(Wu2, 128, 384, 384);
        wmma_gemm_bnact<<<dim3(M1 / 32, 2), 128, 0, stream>>>(
            wpk, bigX, gu2, bu2, d2, nullptr, 384, M1, 1);

        nn1_kernel<<<cdiv(N, 128), 128, 0, stream>>>(xyzb, N, xyz1, M1, nn10);
        concat_interp<<<cdiv(192 * N, 256), 256, 0, stream>>>(d2, nn10, x1, bigX, 128, 64, N, M1);
        pack(Wu1, 64, 192, 192);
        wmma_gemm_bnact<<<dim3(N / 32, 1), 128, 0, stream>>>(
            wpk, bigX, gu1, bu1, nullptr, d1h, 192, N, 1);

        // ---- head ----
        pack(Wh1, 64, 64, 64);
        wmma_gemm_bnact<<<dim3(N / 32, 1), 128, 0, stream>>>(
            wpk, d1h, nullptr, bh1, nullptr, hh, 64, N, 1);
        head2_kernel<<<cdiv(N, 256), 256, 0, stream>>>(hh, Wh2, bh2, outb, N);
    }
}
